// hypercorre_topk2_82008105550129
// MI455X (gfx1250) — compile-verified, matrix operand order checked by reading, not run on hardware
//
#include <hip/hip_runtime.h>
#include <hip/hip_bf16.h>
#include <math.h>

// ---------------------------------------------------------------------------
// Problem constants (B=2, T=3, C=128, H=W=64, NC=150)
// ---------------------------------------------------------------------------
#define BB   2
#define TT   3
#define T1   4
#define GG   8           // B*T1 images
#define CC   128
#define HH   64
#define WW   64
#define NPIX 4096        // H*W
#define NC   150
#define C4   512         // 4*C
#define TOK  600         // T1*NC tokens per batch

typedef __attribute__((ext_vector_type(16))) __bf16 v16bf;
typedef __attribute__((ext_vector_type(8)))  __bf16 v8bf;
typedef __attribute__((ext_vector_type(8)))  float  v8f;
typedef __attribute__((ext_vector_type(4)))  unsigned int u32x4;
typedef __attribute__((ext_vector_type(8)))  int i32x8;
typedef __attribute__((ext_vector_type(4)))  int i32x4;

#if defined(__HIP_DEVICE_COMPILE__) && __has_builtin(__builtin_amdgcn_tensor_load_to_lds) && __has_builtin(__builtin_amdgcn_s_wait_tensorcnt)
#define HAVE_TDM 1
#else
#define HAVE_TDM 0
#endif

#define DEV __device__ __forceinline__

DEV float gelu_exact(float v) {
  return 0.5f * v * (1.0f + erff(v * 0.7071067811865476f));
}

DEV v16bf frag_zero() {
  v16bf f;
#pragma unroll
  for (int e = 0; e < 16; ++e) f[e] = (__bf16)0.0f;
  return f;
}

// ---------------------------------------------------------------------------
// Fragment loaders. CDNA5 16-bit A/B layout (16x32 tile, wave32):
//   lane 0-15  : row/col = lane,    K runs [kb, kb+8) and [kb+16, kb+24)
//   lane 16-31 : row/col = lane-16, kb offset +8
// Caller passes kb = k0 + ((lane>>4)<<3)  (always a multiple of 8).
// ---------------------------------------------------------------------------
DEV v16bf load_frag_rowvec(const float* __restrict__ p, int kb) {
  const float4 a0 = *(const float4*)(p + kb);
  const float4 a1 = *(const float4*)(p + kb + 4);
  const float4 a2 = *(const float4*)(p + kb + 16);
  const float4 a3 = *(const float4*)(p + kb + 20);
  v16bf f;
  f[0] = (__bf16)a0.x; f[1] = (__bf16)a0.y; f[2]  = (__bf16)a0.z; f[3]  = (__bf16)a0.w;
  f[4] = (__bf16)a1.x; f[5] = (__bf16)a1.y; f[6]  = (__bf16)a1.z; f[7]  = (__bf16)a1.w;
  f[8] = (__bf16)a2.x; f[9] = (__bf16)a2.y; f[10] = (__bf16)a2.z; f[11] = (__bf16)a2.w;
  f[12] = (__bf16)a3.x; f[13] = (__bf16)a3.y; f[14] = (__bf16)a3.z; f[15] = (__bf16)a3.w;
  return f;
}

// bf16 source (global or LDS): two 16-byte loads.
DEV v16bf load_frag_bf16vec(const __bf16* p, int kb) {
  v8bf a = *(const v8bf*)(p + kb);
  v8bf b = *(const v8bf*)(p + kb + 16);
  v16bf f;
#pragma unroll
  for (int e = 0; e < 8; ++e) { f[e] = a[e]; f[8 + e] = b[e]; }
  return f;
}

// [K,N] storage (strided along k), unguarded.
DEV v16bf load_frag_kvec(const float* __restrict__ p, int kb, int ld) {
  v16bf f;
#pragma unroll
  for (int e = 0; e < 8; ++e) f[e] = (__bf16)p[(long)(kb + e) * ld];
#pragma unroll
  for (int e = 0; e < 8; ++e) f[8 + e] = (__bf16)p[(long)(kb + 16 + e) * ld];
  return f;
}

// Guarded scalar paths (odd M/N/K only).
DEV v16bf load_frag_rowmajor_g(const float* __restrict__ P, int row, int kb,
                               int R, int K, int ld) {
  v16bf f = frag_zero();
  if (row < R) {
    const float* p = P + (long)row * ld;
#pragma unroll
    for (int e = 0; e < 8; ++e) {
      int k = kb + e;
      if (k < K) f[e] = (__bf16)p[k];
    }
#pragma unroll
    for (int e = 0; e < 8; ++e) {
      int k = kb + 16 + e;
      if (k < K) f[8 + e] = (__bf16)p[k];
    }
  }
  return f;
}

DEV v16bf load_frag_kmajor_g(const float* __restrict__ P, int col, int kb,
                             int N, int K, int ld) {
  v16bf f = frag_zero();
  if (col < N) {
    const float* p = P + col;
#pragma unroll
    for (int e = 0; e < 8; ++e) {
      int k = kb + e;
      if (k < K) f[e] = (__bf16)p[(long)k * ld];
    }
#pragma unroll
    for (int e = 0; e < 8; ++e) {
      int k = kb + 16 + e;
      if (k < K) f[8 + e] = (__bf16)p[(long)k * ld];
    }
  }
  return f;
}

// ---------------------------------------------------------------------------
// Generic WMMA-bf16 GEMM:  D = epi( A[M,K] x B + bias )
//   BLAYOUT: 1 -> B stored [N,K] (weights), 0 -> B stored [K,N] (activations)
//   EPI:     0 none, 1 exact-GELU, 2 sigmoid(beta + alpha*x)
//   TSTORE:  0 -> D[m*N+n], 1 -> D[n*M+m] (transposed, for logits)
//   DMODE:   dest image slot: 0 -> by, 1 -> mem-scatter (by/3)*4+by%3, 2 -> by*4+3
//   GA/GB:   1 -> guarded scalar loads, 0 -> fast vector loads
// ---------------------------------------------------------------------------
template <int BLAYOUT, int EPI, int TSTORE, int DMODE, int GA, int GB>
__global__ void gemm_wmma_kernel(const float* __restrict__ A,
                                 const float* __restrict__ Bm,
                                 const float* __restrict__ bias,
                                 float* __restrict__ D,
                                 int M, int N, int K,
                                 long sA, long sB, long sD,
                                 const float* __restrict__ alpha_p,
                                 const float* __restrict__ beta_p) {
  const int lane = threadIdx.x & 31;
  const int wid  = threadIdx.x >> 5;
  const int tilesN = (N + 15) >> 4;
  const int tilesM = (M + 15) >> 4;
  const int tile = blockIdx.x * 8 + wid;
  if (tile >= tilesM * tilesN) return;
  const int tn = tile % tilesN;
  const int tm = tile / tilesN;
  const int by = blockIdx.y;

  const float* Ab = A + (long)by * sA;
  const float* Bb = Bm + (long)by * sB;
  long dbase;
  if (DMODE == 0)      dbase = (long)by * sD;
  else if (DMODE == 1) dbase = (long)((by / TT) * T1 + (by % TT)) * sD;
  else                 dbase = (long)(by * T1 + (T1 - 1)) * sD;
  float* Db = D + dbase;

  const int rowA  = tm * 16 + (lane & 15);
  const int colB  = tn * 16 + (lane & 15);
  const int khalf = (lane >> 4) << 3;

  const float* arow = Ab + (long)rowA * K;        // valid when GA==0
  const float* brow = Bb + (long)colB * K;        // BLAYOUT==1, GB==0
  const float* bcol = Bb + colB;                  // BLAYOUT==0, GB==0

  v8f acc = {0.f, 0.f, 0.f, 0.f, 0.f, 0.f, 0.f, 0.f};
  for (int k0 = 0; k0 < K; k0 += 32) {
    const int kb = k0 + khalf;
    if (GA == 0 && k0 + 32 < K)   // stream next A chunk (global_prefetch_b8)
      __builtin_prefetch(arow + k0 + 32, 0, 3);
    v16bf a, b;
    if (GA) a = load_frag_rowmajor_g(Ab, rowA, kb, M, K, K);
    else    a = load_frag_rowvec(arow, kb);
    if (BLAYOUT == 1) {
      if (GB) b = load_frag_rowmajor_g(Bb, colB, kb, N, K, K);
      else    b = load_frag_rowvec(brow, kb);
    } else {
      if (GB) b = load_frag_kmajor_g(Bb, colB, kb, N, K, N);
      else    b = load_frag_kvec(bcol, kb, N);
    }
    acc = __builtin_amdgcn_wmma_f32_16x16x32_bf16(false, a, false, b,
                                                  (short)0, acc, false, false);
  }

  float bv = 0.f;
  if (bias != nullptr && colB < N) bv = bias[colB];
  float alpha = 1.f, betav = 0.f;
  if (EPI == 2) { alpha = alpha_p[0]; betav = beta_p[0]; }

#pragma unroll
  for (int r = 0; r < 8; ++r) {
    int m = tm * 16 + r + ((lane >> 4) << 3);
    int n = colB;
    if (m < M && n < N) {
      float v = acc[r] + bv;
      if (EPI == 1)      v = gelu_exact(v);
      else if (EPI == 2) v = 1.f / (1.f + __expf(-(betav + alpha * v)));
      if (TSTORE) Db[(long)n * M + m] = v;
      else        Db[(long)m * N + n] = v;
    }
  }
}

// ---------------------------------------------------------------------------
// Repack 7x7 conv weights: OIHW fp32 [128,128,7,7] -> bf16 [49][out][in]
// ---------------------------------------------------------------------------
__global__ void repack_w7(const float* __restrict__ Wt, __bf16* __restrict__ WB) {
  long idx = (long)blockIdx.x * 256 + threadIdx.x;
  if (idx >= (long)49 * CC * CC) return;
  int i   = (int)(idx & (CC - 1));          // in-channel (k)
  int o   = (int)((idx >> 7) & (CC - 1));   // out-channel (n)
  int tap = (int)(idx >> 14);
  WB[idx] = (__bf16)Wt[((long)o * CC + i) * 49 + tap];
}

// ---------------------------------------------------------------------------
// 7x7 conv C->C (pad 3) over NHWC [G,4096,128] + bias + GELU.
// Block = 256 threads (8 waves) = one 16-pixel strip x all 128 out channels.
//  - A: halo strip (7 x 22 x 128ch) staged once in LDS as bf16, zero-padded.
//  - B: per-tap 128x128 bf16 weight block staged in LDS (TDM when available).
// Per tap: 4 K-steps of v_wmma fed entirely from ds_load_b128.
// ---------------------------------------------------------------------------
#define S_ROWS 7
#define S_COLS 22

__global__ void conv7_gelu_wmma(const float* __restrict__ X,
                                const __bf16* __restrict__ WB,  // [49][128][128]
                                const float* __restrict__ bias,
                                float* __restrict__ Out) {
  __shared__ __bf16 As[S_ROWS * S_COLS * CC];   // 38.5 KB
  __shared__ __bf16 Bs[CC * CC];                // 32 KB

  const int tid  = threadIdx.x;
  const int lane = tid & 31;
  const int wid  = tid >> 5;
  const int tm   = blockIdx.x;     // strip index 0..255
  const int img  = blockIdx.y;

  const int strip = tm * 16;
  const int py0 = strip >> 6;      // image row of the strip
  const int px0 = strip & 63;      // first column (0/16/32/48)

  const float* Xb = X + (long)img * NPIX * CC;

  // ---- stage A halo strip (zero-filled outside the image) ----
  for (int idx = tid; idx < S_ROWS * S_COLS * CC; idx += 256) {
    int ch  = idx & (CC - 1);
    int col = (idx >> 7) % S_COLS;
    int row = (idx >> 7) / S_COLS;
    int sy = py0 + row - 3;
    int sx = px0 + col - 3;
    float v = 0.f;
    if (sy >= 0 && sy < HH && sx >= 0 && sx < WW)
      v = Xb[(long)(sy * WW + sx) * CC + ch];
    As[idx] = (__bf16)v;
  }

  const int ncol  = wid * 16 + (lane & 15);  // this wave's 16 out-channels
  const int c     = lane & 15;               // pixel column within strip
  const int khalf = (lane >> 4) << 3;

  v8f acc = {0.f, 0.f, 0.f, 0.f, 0.f, 0.f, 0.f, 0.f};

  for (int tap = 0; tap < 49; ++tap) {
    __syncthreads();   // A staged (tap 0) / previous tap's compute done
#if HAVE_TDM
    if (wid == 0) {
      // Tensor DMA: copy 32 KB weight block (16384 x bf16, 2D tile) to LDS.
      unsigned long long ga = (unsigned long long)(const void*)(WB + (long)tap * CC * CC);
      unsigned ldsoff = (unsigned)(unsigned long long)(const void*)Bs; // addr[31:0] == LDS offset
      u32x4 g0;
      g0[0] = 1u;                                   // count=1, user descriptor
      g0[1] = ldsoff;                               // lds_addr
      g0[2] = (unsigned)(ga & 0xFFFFFFFFull);       // global_addr[31:0]
      g0[3] = (unsigned)((ga >> 32) & 0x1FFFFFFull) | (2u << 30); // addr[56:32] | type=2
      i32x8 g1;
      g1[0] = (int)(1u << 16);                      // data_size=1 (2 bytes)
      g1[1] = (int)(16384u << 16);                  // tensor_dim0 lo16 -> bits 63:48
      g1[2] = (int)(1u << 16);                      // tensor_dim0 hi16=0 | tensor_dim1 lo16=1
      g1[3] = (int)(16384u << 16);                  // tensor_dim1 hi=0 | tile_dim0=16384
      g1[4] = 1;                                    // tile_dim1=1 | tile_dim2=0
      g1[5] = 16384;                                // tensor_dim0_stride lo32
      g1[6] = 0;
      g1[7] = 0;
      i32x4 gz = {0, 0, 0, 0};
      i32x8 gz8 = {0, 0, 0, 0, 0, 0, 0, 0};
      __builtin_amdgcn_tensor_load_to_lds(g0, g1, gz, gz, gz8, 0);
      __builtin_amdgcn_s_wait_tensorcnt((short)0);
    }
#else
    {
      const v8bf* src = (const v8bf*)(WB + (long)tap * CC * CC);
      v8bf* dst = (v8bf*)Bs;
      for (int i = tid; i < CC * CC / 8; i += 256) dst[i] = src[i];
    }
#endif
    __syncthreads();   // weights visible to all waves

    const int dy = tap / 7;
    const int dx = tap - dy * 7;
    const __bf16* arow = As + ((long)(dy * S_COLS) + (c + dx)) * CC;
    const __bf16* brow = Bs + (long)ncol * CC;
#pragma unroll
    for (int k0 = 0; k0 < CC; k0 += 32) {
      v16bf a = load_frag_bf16vec(arow, k0 + khalf);
      v16bf b = load_frag_bf16vec(brow, k0 + khalf);
      acc = __builtin_amdgcn_wmma_f32_16x16x32_bf16(false, a, false, b,
                                                    (short)0, acc, false, false);
    }
  }

  float bv = bias[ncol];
#pragma unroll
  for (int r = 0; r < 8; ++r) {
    int m = strip + r + ((lane >> 4) << 3);
    Out[(long)img * NPIX * CC + (long)m * CC + ncol] = gelu_exact(acc[r] + bv);
  }
}

// ---------------------------------------------------------------------------
// Elementwise / reduction kernels (memory-bound; VALU)
// ---------------------------------------------------------------------------
__global__ void softmax_rows_4096(float* __restrict__ data) {  // [rows,4096] in place
  const int row = blockIdx.x;
  float* p = data + (long)row * NPIX;
  __shared__ float red[256];
  const int tid = threadIdx.x;
  float mx = -1e30f;
  for (int i = tid; i < NPIX; i += 256) mx = fmaxf(mx, p[i]);
  red[tid] = mx; __syncthreads();
  for (int s = 128; s > 0; s >>= 1) {
    if (tid < s) red[tid] = fmaxf(red[tid], red[tid + s]);
    __syncthreads();
  }
  mx = red[0]; __syncthreads();
  float sum = 0.f;
  for (int i = tid; i < NPIX; i += 256) { float e = __expf(p[i] - mx); p[i] = e; sum += e; }
  red[tid] = sum; __syncthreads();
  for (int s = 128; s > 0; s >>= 1) {
    if (tid < s) red[tid] += red[tid + s];
    __syncthreads();
  }
  float inv = 1.f / red[0];
  for (int i = tid; i < NPIX; i += 256) p[i] *= inv;
}

__global__ void l2norm_rows128(const float* __restrict__ X, float* __restrict__ Y) {
  const int row = blockIdx.x;
  const int c = threadIdx.x;  // 128 threads
  float v = X[(long)row * CC + c];
  __shared__ float red[128];
  red[c] = v * v; __syncthreads();
  for (int s = 64; s > 0; s >>= 1) { if (c < s) red[c] += red[c + s]; __syncthreads(); }
  float inv = 1.f / fmaxf(sqrtf(red[0]), 1e-12f);
  Y[(long)row * CC + c] = v * inv;
}

// y = z * clip(dot(l2norm(z_row), l2norm(prompt)), 0, 1)
__global__ void prompt_gate(const float* __restrict__ Z, const float* __restrict__ prompt,
                            float* __restrict__ Y) {
  const int row = blockIdx.x;
  const int c = threadIdx.x;  // 128 threads
  __shared__ float red[128];
  __shared__ float pred[128];
  float zv = Z[(long)row * CC + c];
  float pv = prompt[c];
  red[c] = zv * zv; pred[c] = pv * pv; __syncthreads();
  for (int s = 64; s > 0; s >>= 1) {
    if (c < s) { red[c] += red[c + s]; pred[c] += pred[c + s]; }
    __syncthreads();
  }
  float zinv = 1.f / fmaxf(sqrtf(red[0]), 1e-12f);
  float pinv = 1.f / fmaxf(sqrtf(pred[0]), 1e-12f);
  __syncthreads();
  red[c] = (zv * zinv) * (pv * pinv); __syncthreads();
  for (int s = 64; s > 0; s >>= 1) { if (c < s) red[c] += red[c + s]; __syncthreads(); }
  float m = fminf(fmaxf(red[0], 0.f), 1.f);
  Y[(long)row * CC + c] = zv * m;
}

__global__ void avg_kernel(const float* __restrict__ a, const float* __restrict__ b,
                           float* __restrict__ y, int n) {
  int i = blockIdx.x * 256 + threadIdx.x;
  if (i < n) y[i] = 0.5f * (a[i] + b[i]);
}

// y = res + LayerNorm(x)*g + b   (rows of 128)
__global__ void ln_residual(const float* __restrict__ res, const float* __restrict__ x,
                            const float* __restrict__ g, const float* __restrict__ bb,
                            float* __restrict__ y) {
  const int row = blockIdx.x;
  const int c = threadIdx.x;  // 128 threads
  float v = x[(long)row * CC + c];
  __shared__ float red[128];
  red[c] = v; __syncthreads();
  for (int s = 64; s > 0; s >>= 1) { if (c < s) red[c] += red[c + s]; __syncthreads(); }
  float mean = red[0] * (1.f / CC);
  __syncthreads();
  float d = v - mean;
  red[c] = d * d; __syncthreads();
  for (int s = 64; s > 0; s >>= 1) { if (c < s) red[c] += red[c + s]; __syncthreads(); }
  float var = red[0] * (1.f / CC);
  float nv = d * rsqrtf(var + 1e-5f);
  y[(long)row * CC + c] = res[(long)row * CC + c] + nv * g[c] + bb[c];
}

// depthwise 3x3 (pad 1, groups=512) on [2,4096,512] NHWC + bias + GELU
__global__ void dwconv3_gelu(const float* __restrict__ X, const float* __restrict__ Wd,
                             const float* __restrict__ bd, float* __restrict__ Y) {
  long i = (long)blockIdx.x * 256 + threadIdx.x;
  if (i >= (long)BB * NPIX * C4) return;
  int ch = (int)(i & (C4 - 1));
  int p  = (int)((i >> 9) & (NPIX - 1));
  int bi = (int)(i >> 21);
  int py = p >> 6, px = p & 63;
  const float* base = X + (long)bi * NPIX * C4;
  float acc = bd[ch];
#pragma unroll
  for (int dy = 0; dy < 3; ++dy) {
    int sy = py + dy - 1;
#pragma unroll
    for (int dx = 0; dx < 3; ++dx) {
      int sx = px + dx - 1;
      if (sy >= 0 && sy < HH && sx >= 0 && sx < WW)
        acc += base[(long)(sy * WW + sx) * C4 + ch] * Wd[ch * 9 + dy * 3 + dx];
    }
  }
  Y[i] = gelu_exact(acc);
}

// ---------------------------------------------------------------------------
// Host orchestration
// ---------------------------------------------------------------------------
static inline int cdiv(int a, int b) { return (a + b - 1) / b; }

extern "C" void kernel_launch(void* const* d_in, const int* in_sizes, int n_in,
                              void* d_out, int out_size, void* d_ws, size_t ws_size,
                              hipStream_t stream) {
  (void)in_sizes; (void)n_in; (void)out_size; (void)ws_size;
  const float* x    = (const float*)d_in[0];   // [2,4096,128]
  const float* z    = (const float*)d_in[1];   // [2,600,128]
  const float* mem  = (const float*)d_in[2];   // [6,4096,128]
  const float* csw  = (const float*)d_in[3];   // [128,128]
  const float* csb  = (const float*)d_in[4];
  const float* c1w  = (const float*)d_in[5];
  const float* c1b  = (const float*)d_in[6];
  const float* kw1  = (const float*)d_in[7];   // [128,128,7,7]
  const float* kb1  = (const float*)d_in[8];
  const float* kw2  = (const float*)d_in[9];
  const float* kb2  = (const float*)d_in[10];
  const float* kw3  = (const float*)d_in[11];  // [150,128]
  const float* pr1  = (const float*)d_in[12];
  const float* tdt1 = (const float*)d_in[13];
  const float* pr2  = (const float*)d_in[14];
  const float* tdt2 = (const float*)d_in[15];
  const float* s_a  = (const float*)d_in[16];
  const float* s_b  = (const float*)d_in[17];
  const float* lng  = (const float*)d_in[18];
  const float* lnb  = (const float*)d_in[19];
  const float* f1w  = (const float*)d_in[20];  // [512,128]
  const float* f1b  = (const float*)d_in[21];
  const float* dww  = (const float*)d_in[22];  // [512,9]
  const float* dwb  = (const float*)d_in[23];
  const float* f2w  = (const float*)d_in[24];  // [128,512]
  const float* f2b  = (const float*)d_in[25];

  float* out_final  = (float*)d_out;                       // [2,4096,128]
  float* out_center = out_final + (long)BB * NPIX * CC;    // [8,150,128]
  float* out_cin    = out_center + (long)GG * NC * CC;     // [2,600,128]

  float* f = (float*)d_ws;
  float* XA   = f;                             // [8,4096,128]
  float* FEAT = XA   + (long)GG * NPIX * CC;   // [2,4096,128]
  float* Hb   = FEAT + (long)BB * NPIX * CC;   // [8,4096,128]
  float* H2   = Hb   + (long)GG * NPIX * CC;   // [8,4096,128]
  float* ASG  = H2   + (long)GG * NPIX * CC;   // [8,150,4096]
  float* ZG   = ASG  + (long)GG * NC * NPIX;   // [2,600,128]
  float* ZT   = ZG   + (long)BB * TOK * CC;
  float* CZG  = ZT   + (long)BB * TOK * CC;
  float* CZT  = CZG  + (long)BB * TOK * CC;
  float* FHAT = CZT  + (long)BB * TOK * CC;    // [2,4096,128]
  float* CHAT = FHAT + (long)BB * NPIX * CC;   // [2,600,128]
  float* SIM  = CHAT + (long)BB * TOK * CC;    // [2,4096,600]
  float* ATTN = SIM  + (long)BB * NPIX * TOK;  // [2,4096,128]
  float* OUT1 = ATTN + (long)BB * NPIX * CC;   // [2,4096,128]
  float* H1   = OUT1 + (long)BB * NPIX * CC;   // [2,4096,512]
  float* H1G  = H1   + (long)BB * NPIX * C4;   // [2,4096,512]
  float* MLP  = H1G  + (long)BB * NPIX * C4;   // [2,4096,128]
  __bf16* WB1 = (__bf16*)(MLP + (long)BB * NPIX * CC); // [49,128,128] bf16

  dim3 blk(256);

  // 0) repack 7x7 weights to bf16 [tap][out][in]
  repack_w7<<<dim3(cdiv(49 * CC * CC, 256)), blk, 0, stream>>>(kw1, WB1);

  // 1) feature = conv_skip(x)           [8192,128] = [8192,128] x [128,128]^T
  gemm_wmma_kernel<1, 0, 0, 0, 0, 0><<<dim3(cdiv(512 * 8, 8), 1), blk, 0, stream>>>(
      x, csw, csb, FEAT, BB * NPIX, CC, CC, 0, 0, 0, nullptr, nullptr);

  // 2) conv1(x) -> XA slots (b*4+3)
  gemm_wmma_kernel<1, 0, 0, 2, 0, 0><<<dim3(cdiv(256 * 8, 8), BB), blk, 0, stream>>>(
      x, c1w, c1b, XA, NPIX, CC, CC, (long)NPIX * CC, 0, (long)NPIX * CC,
      nullptr, nullptr);

  // 3) conv1(mem) -> XA slots (b*4 + t)
  gemm_wmma_kernel<1, 0, 0, 1, 0, 0><<<dim3(cdiv(256 * 8, 8), BB * TT), blk, 0, stream>>>(
      mem, c1w, c1b, XA, NPIX, CC, CC, (long)NPIX * CC, 0, (long)NPIX * CC,
      nullptr, nullptr);

  // 4) H = gelu(conv7x7(XA))  -- LDS-staged implicit GEMM
  conv7_gelu_wmma<<<dim3(256, GG), blk, 0, stream>>>(XA, WB1, kb1, Hb);

  // 5) H2 = gelu(conv1x1(H))            M=32768
  gemm_wmma_kernel<1, 1, 0, 0, 0, 0><<<dim3(cdiv(2048 * 8, 8), 1), blk, 0, stream>>>(
      Hb, kw2, kb2, H2, GG * NPIX, CC, CC, 0, 0, 0, nullptr, nullptr);

  // 6) logits = conv1x1(H2) (no bias), transposed store -> [8,150,4096]
  gemm_wmma_kernel<1, 0, 1, 0, 0, 1><<<dim3(cdiv(256 * 10, 8), GG), blk, 0, stream>>>(
      H2, kw3, nullptr, ASG, NPIX, NC, CC, (long)NPIX * CC, 0, (long)NC * NPIX,
      nullptr, nullptr);

  // 7) softmax over 4096 pixels per (img, cluster) row
  softmax_rows_4096<<<dim3(GG * NC), blk, 0, stream>>>(ASG);

  // 8) centers = assign @ xa -> d_out  [8,150,128], K=4096
  gemm_wmma_kernel<0, 0, 0, 0, 1, 0><<<dim3(cdiv(10 * 8, 8), GG), blk, 0, stream>>>(
      ASG, XA, nullptr, out_center, NC, CC, NPIX,
      (long)NC * NPIX, (long)NPIX * CC, (long)NC * CC, nullptr, nullptr);

  // 9-12) prompt gating + top-down transforms
  prompt_gate<<<dim3(BB * TOK), dim3(128), 0, stream>>>(z, pr1, ZG);
  gemm_wmma_kernel<0, 0, 0, 0, 0, 0><<<dim3(cdiv(75 * 8, 8), 1), blk, 0, stream>>>(
      ZG, tdt1, nullptr, ZT, BB * TOK, CC, CC, 0, 0, 0, nullptr, nullptr);
  prompt_gate<<<dim3(BB * TOK), dim3(128), 0, stream>>>(out_center, pr2, CZG);
  gemm_wmma_kernel<0, 0, 0, 0, 0, 0><<<dim3(cdiv(75 * 8, 8), 1), blk, 0, stream>>>(
      CZG, tdt2, nullptr, CZT, BB * TOK, CC, CC, 0, 0, 0, nullptr, nullptr);

  // 13) C_in = (cz_t + z_t)/2 -> d_out
  avg_kernel<<<dim3(cdiv(BB * TOK * CC, 256)), blk, 0, stream>>>(
      CZT, ZT, out_cin, BB * TOK * CC);

  // 14-15) l2-normalize rows for cosine sim
  l2norm_rows128<<<dim3(BB * NPIX), dim3(128), 0, stream>>>(FEAT, FHAT);
  l2norm_rows128<<<dim3(BB * TOK), dim3(128), 0, stream>>>(out_cin, CHAT);

  // 16) similarity = sigmoid(beta + alpha * fhat @ chat^T)   [2,4096,600]
  gemm_wmma_kernel<1, 2, 0, 0, 0, 1><<<dim3(cdiv(256 * 38, 8), BB), blk, 0, stream>>>(
      FHAT, CHAT, nullptr, SIM, NPIX, TOK, CC,
      (long)NPIX * CC, (long)TOK * CC, (long)NPIX * TOK, s_a, s_b);

  // 17) attn = similarity @ C_in        [2,4096,128], K=600
  gemm_wmma_kernel<0, 0, 0, 0, 1, 1><<<dim3(cdiv(256 * 8, 8), BB), blk, 0, stream>>>(
      SIM, out_cin, nullptr, ATTN, NPIX, CC, TOK,
      (long)NPIX * TOK, (long)TOK * CC, (long)NPIX * CC, nullptr, nullptr);

  // 18) out1 = x + LN(attn)
  ln_residual<<<dim3(BB * NPIX), dim3(128), 0, stream>>>(x, ATTN, lng, lnb, OUT1);

  // 19) h1 = out1 @ fc1_w^T + b         [8192,512]
  gemm_wmma_kernel<1, 0, 0, 0, 0, 0><<<dim3(cdiv(512 * 32, 8), 1), blk, 0, stream>>>(
      OUT1, f1w, f1b, H1, BB * NPIX, C4, CC, 0, 0, 0, nullptr, nullptr);

  // 20) depthwise 3x3 + GELU
  dwconv3_gelu<<<dim3(cdiv(BB * NPIX * C4, 256)), blk, 0, stream>>>(H1, dww, dwb, H1G);

  // 21) mlp = h1g @ fc2_w^T + b         [8192,128], K=512
  gemm_wmma_kernel<1, 0, 0, 0, 0, 0><<<dim3(cdiv(512 * 8, 8), 1), blk, 0, stream>>>(
      H1G, f2w, f2b, MLP, BB * NPIX, CC, C4, 0, 0, 0, nullptr, nullptr);

  // 22) out = out1 + LN(mlp) -> d_out
  ln_residual<<<dim3(BB * NPIX), dim3(128), 0, stream>>>(OUT1, MLP, lng, lnb, out_final);
}
